// SAGEMLP_60971355734529
// MI455X (gfx1250) — matrix operand order repaired and checked by reading, not compiled, not run on hardware
//
#include <hip/hip_runtime.h>
#include <hip/hip_bf16.h>
#include <math.h>

typedef __attribute__((ext_vector_type(16))) _Float16 v16h;
typedef __attribute__((ext_vector_type(8)))  _Float16 v8h;
typedef __attribute__((ext_vector_type(8)))  float    v8f;

#define SAGE_N 50000   // multiple of 16 -> all WMMA row tiles are full
#define SAGE_E 800000
#define SAGE_G 64

// ---------------------------------------------------------------------------
// helpers
// ---------------------------------------------------------------------------
__device__ __forceinline__ float gelu_exact(float x) {
  return 0.5f * x * (1.0f + erff(x * 0.7071067811865475f));
}

__device__ __forceinline__ float wave_sum32(float v) {
#pragma unroll
  for (int off = 16; off > 0; off >>= 1) v += __shfl_xor(v, off, 32);
  return v;
}

// ---------------------------------------------------------------------------
// zero fill
// ---------------------------------------------------------------------------
__global__ void sage_zero_f32(float* __restrict__ p, long n) {
  long i = (long)blockIdx.x * blockDim.x + threadIdx.x;
  if (i < n) p[i] = 0.0f;
}

// ---------------------------------------------------------------------------
// degree: atomicAdd 1 per edge destination
// ---------------------------------------------------------------------------
__global__ void sage_deg(const int* __restrict__ dst, float* __restrict__ deg, int E) {
  int e = blockIdx.x * blockDim.x + threadIdx.x;
  if (e < E) atomicAdd(&deg[dst[e]], 1.0f);
}

// ---------------------------------------------------------------------------
// x [N,K] f32 -> xh [N,Kpad] f16 (zero pad K tail)
// ---------------------------------------------------------------------------
__global__ void sage_x2h(const float* __restrict__ x, _Float16* __restrict__ xh,
                         int N, int K, int Kpad) {
  long t = (long)blockIdx.x * blockDim.x + threadIdx.x;
  if (t >= (long)N * Kpad) return;
  int n = (int)(t / Kpad);
  int c = (int)(t % Kpad);
  xh[t] = (c < K) ? (_Float16)x[(long)n * K + c] : (_Float16)0.0f;
}

// ---------------------------------------------------------------------------
// Repack f32 weight [nmat][K,64] into WMMA B-fragment order (f16):
// frag[m][(kt*4+nt)*32 + lane][h]  where h=0..15, so each lane's 16 halves
// are 32 contiguous bytes -> one v16h load (2x global_load_b128, coalesced).
// ISA 16-bit B layout: col = nt*16 + (lane&15); k = kt*32 + ((lane&16)?8:0) + (h&7) + ((h&8)?16:0).
// ---------------------------------------------------------------------------
__global__ void sage_wfrag(const float* __restrict__ W, _Float16* __restrict__ frag,
                           int K, int KT, int nmat, long wstride, long fstride) {
  long t = (long)blockIdx.x * blockDim.x + threadIdx.x;
  long per = (long)KT * 2048;  // KT*4 tiles * 32 lanes * 16 halves
  if (t >= (long)nmat * per) return;
  int  m = (int)(t / per);
  long r = t % per;
  int h    = (int)(r & 15);
  int lane = (int)((r >> 4) & 31);
  int tile = (int)(r >> 9);
  int nt = tile & 3;
  int kt = tile >> 2;
  int col = nt * 16 + (lane & 15);
  int k   = kt * 32 + ((lane & 16) ? 8 : 0) + (h & 7) + ((h & 8) ? 16 : 0);
  frag[m * fstride + r] =
      (k < K) ? (_Float16)W[m * wstride + (long)k * 64 + col] : (_Float16)0.0f;
}

// ---------------------------------------------------------------------------
// Dual GEMM via WMMA: Yl = X @ Wl, Yr = X @ Wr.
// Xh: [N,KT*32] f16 (padded).  WAf/WBf: fragment-packed f16 weights.
// One wave -> full 16 rows x 64 cols for BOTH outputs (shared A fragment).
// All 8 B fragments of a k-tile are loaded before the 8-WMMA burst so a
// single loadcnt wait covers the clause and the next tile's loads overlap.
// ---------------------------------------------------------------------------
#define GEMM_WAVES 4
template <int KT>
__global__ void sage_gemm_dual(const _Float16* __restrict__ Xh, int N,
                               const _Float16* __restrict__ WAf,
                               const _Float16* __restrict__ WBf,
                               float* __restrict__ YA, float* __restrict__ YB) {
  const int lane = threadIdx.x & 31;
  const int wid  = threadIdx.x >> 5;
  const int rowBase = (blockIdx.x * GEMM_WAVES + wid) * 16;
  if (rowBase >= N) return;  // wave-uniform; N%16==0 so tiles below are full

  const int mrow = rowBase + (lane & 15);
  const int kbA  = (lane & 16) ? 8 : 0;  // ISA 16-bit A layout half-offset

  const v16h* __restrict__ fA = (const v16h*)WAf;
  const v16h* __restrict__ fB = (const v16h*)WBf;
  // per-lane A stream: two aligned 16B runs per 32-k tile
  const v8h* __restrict__ xrow = (const v8h*)(Xh + (long)mrow * (KT * 32) + kbA);

  v8f accA[4], accB[4];
#pragma unroll
  for (int nt = 0; nt < 4; ++nt) {
    accA[nt] = (v8f){0,0,0,0,0,0,0,0};
    accB[nt] = (v8f){0,0,0,0,0,0,0,0};
  }

#pragma unroll
  for (int kt = 0; kt < KT; ++kt) {
    const v8h lo = xrow[kt * 4 + 0];  // K = kt*32 + kbA + {0..7}
    const v8h hi = xrow[kt * 4 + 2];  // K = kt*32 + kbA + {16..23}
    v16h a;
#pragma unroll
    for (int h = 0; h < 8; ++h) { a[h] = lo[h]; a[h + 8] = hi[h]; }

    const int tb = kt * 4 * 32 + lane;
    v16h bA[4], bB[4];
#pragma unroll
    for (int nt = 0; nt < 4; ++nt) {
      bA[nt] = fA[tb + nt * 32];
      bB[nt] = fB[tb + nt * 32];
    }
#pragma unroll
    for (int nt = 0; nt < 4; ++nt) {
      accA[nt] = __builtin_amdgcn_wmma_f32_16x16x32_f16(
          false, a, false, bA[nt], (short)0, accA[nt], false, false);
      accB[nt] = __builtin_amdgcn_wmma_f32_16x16x32_f16(
          false, a, false, bB[nt], (short)0, accB[nt], false, false);
    }
  }

  const int drow0 = rowBase + ((lane & 16) ? 8 : 0);
  const int col = lane & 15;
#pragma unroll
  for (int r = 0; r < 8; ++r) {
    const long rb = (long)(drow0 + r) * 64 + col;
#pragma unroll
    for (int nt = 0; nt < 4; ++nt) {
      YA[rb + nt * 16] = accA[nt][r];
      YB[rb + nt * 16] = accB[nt][r];
    }
  }
}

// ---------------------------------------------------------------------------
// edge scatter-add: msum[dst] += Yl[src]   (64-dim payload, 4 dims/thread)
// ---------------------------------------------------------------------------
__global__ void sage_scatter(const int* __restrict__ src, const int* __restrict__ dst,
                             const float* __restrict__ Y, float* __restrict__ msum, int E) {
  long t = (long)blockIdx.x * blockDim.x + threadIdx.x;
  if (t >= (long)E * 16) return;
  int e = (int)(t >> 4);
  int d = (int)(t & 15) * 4;
  const float* ys = Y + (long)src[e] * 64 + d;
  float* md = msum + (long)dst[e] * 64 + d;
  atomicAdd(md + 0, ys[0]);
  atomicAdd(md + 1, ys[1]);
  atomicAdd(md + 2, ys[2]);
  atomicAdd(md + 3, ys[3]);
}

// ---------------------------------------------------------------------------
// fused mean / bias / lin_r add / GELU / LayerNorm / residual.  1 wave = 1 node.
// Writes f32 state (residual/pool) AND f16 copy (next GEMM's A operand).
// ---------------------------------------------------------------------------
__global__ void sage_combine(const float* __restrict__ msum, const float* __restrict__ deg,
                             const float* __restrict__ yr, const float* __restrict__ blv,
                             const float* __restrict__ gam, const float* __restrict__ bet,
                             const float* __restrict__ hprev, float* __restrict__ hout,
                             _Float16* __restrict__ hhout, int N, int residual) {
  const int lane = threadIdx.x & 31;
  const int node = blockIdx.x * (blockDim.x >> 5) + (threadIdx.x >> 5);
  if (node >= N) return;
  const float inv = 1.0f / fmaxf(deg[node], 1.0f);
  const long base = (long)node * 64;
  float v0 = msum[base + lane]      * inv + blv[lane]      + yr[base + lane];
  float v1 = msum[base + lane + 32] * inv + blv[lane + 32] + yr[base + lane + 32];
  v0 = gelu_exact(v0);
  v1 = gelu_exact(v1);
  const float mu = wave_sum32(v0 + v1) * (1.0f / 64.0f);
  const float d0 = v0 - mu, d1 = v1 - mu;
  const float var = wave_sum32(d0 * d0 + d1 * d1) * (1.0f / 64.0f);
  const float rstd = rsqrtf(var + 1e-5f);
  float o0 = d0 * rstd * gam[lane]      + bet[lane];
  float o1 = d1 * rstd * gam[lane + 32] + bet[lane + 32];
  if (residual) {
    o0 += hprev[base + lane];
    o1 += hprev[base + lane + 32];
  }
  hout[base + lane]       = o0;
  hout[base + lane + 32]  = o1;
  hhout[base + lane]      = (_Float16)o0;
  hhout[base + lane + 32] = (_Float16)o1;
}

// ---------------------------------------------------------------------------
// pool: pooled[batch[i]] += h[i]
// ---------------------------------------------------------------------------
__global__ void sage_pool(const float* __restrict__ h, const int* __restrict__ batch,
                          float* __restrict__ pooled, int N) {
  const int lane = threadIdx.x & 31;
  const int node = blockIdx.x * (blockDim.x >> 5) + (threadIdx.x >> 5);
  if (node >= N) return;
  const int gidx = batch[node];
  const long base = (long)node * 64;
  atomicAdd(&pooled[gidx * 64 + lane],      h[base + lane]);
  atomicAdd(&pooled[gidx * 64 + lane + 32], h[base + lane + 32]);
}

// ---------------------------------------------------------------------------
// tiny MLP head: 64 blocks (one per graph) x 32 threads
// ---------------------------------------------------------------------------
__global__ void sage_mlp(const float* __restrict__ pooled,
                         const float* __restrict__ M0, const float* __restrict__ mb0,
                         const float* __restrict__ mg0, const float* __restrict__ mbeta0,
                         const float* __restrict__ M,  const float* __restrict__ mb,
                         const float* __restrict__ mg, const float* __restrict__ mbeta,
                         const float* __restrict__ Wf, const float* __restrict__ bf,
                         float* __restrict__ out) {
  __shared__ float p[64];
  __shared__ float m[32];
  const int gi = blockIdx.x;
  const int l  = threadIdx.x;  // 0..31
  p[l]      = pooled[gi * 64 + l];
  p[l + 32] = pooled[gi * 64 + l + 32];
  __syncthreads();

  float acc = mb0[l];
  for (int k = 0; k < 64; ++k) acc += p[k] * M0[k * 32 + l];
  acc = gelu_exact(acc);
  float mu  = wave_sum32(acc) * (1.0f / 32.0f);
  float d   = acc - mu;
  float var = wave_sum32(d * d) * (1.0f / 32.0f);
  float cur = d * rsqrtf(var + 1e-5f) * mg0[l] + mbeta0[l];

  for (int i = 0; i < 3; ++i) {
    m[l] = cur;
    __syncthreads();
    float a2 = mb[i * 32 + l];
    const float* Mi = M + i * 32 * 32;
    for (int k = 0; k < 32; ++k) a2 += m[k] * Mi[k * 32 + l];
    a2 = gelu_exact(a2);
    float mu2  = wave_sum32(a2) * (1.0f / 32.0f);
    float d2   = a2 - mu2;
    float var2 = wave_sum32(d2 * d2) * (1.0f / 32.0f);
    cur = d2 * rsqrtf(var2 + 1e-5f) * mg[i * 32 + l] + mbeta[i * 32 + l] + cur;
    __syncthreads();
  }

  float f = wave_sum32(cur * Wf[l]);
  if (l == 0) out[gi] = f + bf[0];
}

// ---------------------------------------------------------------------------
// host launch
// ---------------------------------------------------------------------------
extern "C" void kernel_launch(void* const* d_in, const int* in_sizes, int n_in,
                              void* d_out, int out_size, void* d_ws, size_t ws_size,
                              hipStream_t stream) {
  const float* x     = (const float*)d_in[0];
  const int*   eidx  = (const int*)d_in[1];
  const int*   batch = (const int*)d_in[2];
  const float* Wl0   = (const float*)d_in[3];
  const float* bl0   = (const float*)d_in[4];
  const float* Wr0   = (const float*)d_in[5];
  const float* g0    = (const float*)d_in[6];
  const float* beta0 = (const float*)d_in[7];
  const float* Wl    = (const float*)d_in[8];
  const float* bl    = (const float*)d_in[9];
  const float* Wr    = (const float*)d_in[10];
  const float* g     = (const float*)d_in[11];
  const float* beta  = (const float*)d_in[12];
  const float* M0    = (const float*)d_in[13];
  const float* mb0   = (const float*)d_in[14];
  const float* mg0   = (const float*)d_in[15];
  const float* mbeta0= (const float*)d_in[16];
  const float* M     = (const float*)d_in[17];
  const float* mb    = (const float*)d_in[18];
  const float* mg    = (const float*)d_in[19];
  const float* mbeta = (const float*)d_in[20];
  const float* Wf    = (const float*)d_in[21];
  const float* bf    = (const float*)d_in[22];
  float* out = (float*)d_out;

  const int N = SAGE_N, E = SAGE_E;
  const int D_IN = 261, KPAD0 = 288;  // 261 padded to 9*32
  const int* src = eidx;
  const int* dst = eidx + E;

  // workspace bump allocator
  char* ws = (char*)d_ws;
  size_t off = 0;
  auto balloc = [&](size_t bytes) -> void* {
    void* p = ws + off;
    off += (bytes + 255) & ~(size_t)255;
    return p;
  };
  float*    deg    = (float*)balloc((size_t)N * 4);
  float*    hA     = (float*)balloc((size_t)N * 64 * 4);
  float*    hB     = (float*)balloc((size_t)N * 64 * 4);
  _Float16* hhA    = (_Float16*)balloc((size_t)N * 64 * 2);
  _Float16* hhB    = (_Float16*)balloc((size_t)N * 64 * 2);
  float*    yl     = (float*)balloc((size_t)N * 64 * 4);
  float*    yr     = (float*)balloc((size_t)N * 64 * 4);
  float*    msum   = (float*)balloc((size_t)N * 64 * 4);
  float*    pooled = (float*)balloc(64 * 64 * 4);
  _Float16* xh     = (_Float16*)balloc((size_t)N * KPAD0 * 2);
  _Float16* wl0f   = (_Float16*)balloc((size_t)9 * 2048 * 2);   // KT0=9 frags
  _Float16* wr0f   = (_Float16*)balloc((size_t)9 * 2048 * 2);
  _Float16* wlf    = (_Float16*)balloc((size_t)6 * 2 * 2048 * 2);  // 6 layers, KT=2
  _Float16* wrf    = (_Float16*)balloc((size_t)6 * 2 * 2048 * 2);

  // --- degree (reused by all 7 blocks) ---
  sage_zero_f32<<<(N + 255) / 256, 256, 0, stream>>>(deg, N);
  sage_deg<<<(E + 255) / 256, 256, 0, stream>>>(dst, deg, E);

  // --- activations and weights -> f16 / fragment-packed ---
  const long nxh = (long)N * KPAD0;
  sage_x2h<<<(int)((nxh + 255) / 256), 256, 0, stream>>>(x, xh, N, D_IN, KPAD0);
  sage_wfrag<<<(9 * 2048 + 255) / 256, 256, 0, stream>>>(Wl0, wl0f, D_IN, 9, 1, 0, 0);
  sage_wfrag<<<(9 * 2048 + 255) / 256, 256, 0, stream>>>(Wr0, wr0f, D_IN, 9, 1, 0, 0);
  sage_wfrag<<<(6 * 2 * 2048 + 255) / 256, 256, 0, stream>>>(Wl, wlf, 64, 2, 6,
                                                             64 * 64, 2 * 2048);
  sage_wfrag<<<(6 * 2 * 2048 + 255) / 256, 256, 0, stream>>>(Wr, wrf, 64, 2, 6,
                                                             64 * 64, 2 * 2048);

  const int gemmGrid = (N + 16 * GEMM_WAVES - 1) / (16 * GEMM_WAVES);
  const long nh = (long)N * 64;
  const int zeroGrid = (int)((nh + 255) / 256);
  const int scGrid = (int)(((long)E * 16 + 255) / 256);
  const int nodeGrid = (N + 7) / 8;  // 8 waves/block, 1 wave/node

  // --- block 0: 261 -> 64 (aggregate AFTER the linear map: linearity of mean) ---
  sage_gemm_dual<9><<<gemmGrid, 32 * GEMM_WAVES, 0, stream>>>(xh, N, wl0f, wr0f, yl, yr);
  sage_zero_f32<<<zeroGrid, 256, 0, stream>>>(msum, nh);
  sage_scatter<<<scGrid, 256, 0, stream>>>(src, dst, yl, msum, E);
  sage_combine<<<nodeGrid, 256, 0, stream>>>(msum, deg, yr, bl0, g0, beta0,
                                             yl /*unused*/, hA, hhA, N, 0);

  // --- 6 residual SAGE blocks 64 -> 64 ---
  float* hin = hA;   float* hout = hB;
  _Float16* hhin = hhA; _Float16* hhout = hhB;
  for (int i = 0; i < 6; ++i) {
    sage_gemm_dual<2><<<gemmGrid, 32 * GEMM_WAVES, 0, stream>>>(
        hhin, N, wlf + (size_t)i * 2 * 2048, wrf + (size_t)i * 2 * 2048, yl, yr);
    sage_zero_f32<<<zeroGrid, 256, 0, stream>>>(msum, nh);
    sage_scatter<<<scGrid, 256, 0, stream>>>(src, dst, yl, msum, E);
    sage_combine<<<nodeGrid, 256, 0, stream>>>(msum, deg, yr, bl + i * 64,
                                               g + i * 64, beta + i * 64,
                                               hin, hout, hhout, N, 1);
    float* t = hin; hin = hout; hout = t;
    _Float16* th = hhin; hhin = hhout; hhout = th;
  }

  // --- pooling + MLP head ---
  sage_zero_f32<<<(64 * 64 + 255) / 256, 256, 0, stream>>>(pooled, 64 * 64);
  sage_pool<<<nodeGrid, 256, 0, stream>>>(hin, batch, pooled, N);
  sage_mlp<<<SAGE_G, 32, 0, stream>>>(pooled, M0, mb0, mg0, mbeta0,
                                      M, mb, mg, mbeta, Wf, bf, out);
}
